// Attention_67242007987032
// MI455X (gfx1250) — matrix operand order repaired and checked by reading, not compiled, not run on hardware
//
#include <hip/hip_runtime.h>

#define DIM   128
#define HEADS 4
#define NTOK  196
#define NQP   208   // 13*16 padded tokens (QK columns / q rows)
#define NKP   224   // 7*32 padded K for PV
#define HD    32
#define BATCH 1024
#define NWIN  64
#define LBIAS 729
#define PDIM  8
#define LNEPS 1e-5f
#define SCALE 0.17677669529663687f  // 32^-0.5

typedef __attribute__((ext_vector_type(16))) _Float16 v16h;
typedef __attribute__((ext_vector_type(8)))  float    v8f;
typedef __attribute__((ext_vector_type(4)))  unsigned int v4u;
typedef __attribute__((ext_vector_type(4)))  float    v4f;

union Frag { v16h v; v4u u[2]; };

__device__ __forceinline__ v8f wmma_f16(const Frag& a, const Frag& b, v8f c) {
  return __builtin_amdgcn_wmma_f32_16x16x32_f16(false, a.v, false, b.v, (short)0, c, false, false);
}

// ---------------- workspace layout (bytes, 256-aligned) ----------------
static constexpr size_t OFF_PTAB  = 0;                                        // 729*4 f32
static constexpr size_t OFF_RPBT  = 12288;                                    // 4*196*208 f32 (pad cols = -1e30)
static constexpr size_t OFF_MASKP = OFF_RPBT + (size_t)HEADS*NTOK*NQP*4;      // 64*196*208 f32 (pad cols = 0)
static constexpr size_t OFF_WQKV  = OFF_MASKP + (size_t)NWIN*NTOK*NQP*4;      // 384*128 f16 (transposed)
static constexpr size_t OFF_WPROJ = OFF_WQKV + (size_t)384*128*2;             // 128*128 f16 (transposed)
static constexpr size_t OFF_QF    = OFF_WPROJ + (size_t)128*128*2;            // [B][H][208][32] f16
static constexpr size_t OFF_KF    = OFF_QF + (size_t)BATCH*HEADS*NQP*HD*2;    // [B][H][208][32] f16
static constexpr size_t OFF_VT    = OFF_KF + (size_t)BATCH*HEADS*NQP*HD*2;    // [B][H][32][224] f16

// ======================= 1) DynamicPosBias MLP =======================
__device__ __forceinline__ void ln_relu(const float* v, float* t, const float* w, const float* b) {
  float mu = 0.f;
  #pragma unroll
  for (int d = 0; d < PDIM; ++d) mu += v[d];
  mu *= (1.f / PDIM);
  float var = 0.f;
  #pragma unroll
  for (int d = 0; d < PDIM; ++d) { float z = v[d] - mu; var += z * z; }
  var *= (1.f / PDIM);
  float r = rsqrtf(var + LNEPS);
  #pragma unroll
  for (int d = 0; d < PDIM; ++d) {
    float y = (v[d] - mu) * r * w[d] + b[d];
    t[d] = y > 0.f ? y : 0.f;
  }
}

__global__ void pos_mlp_kernel(const float* __restrict__ biases,
    const float* __restrict__ pw,  const float* __restrict__ pb,
    const float* __restrict__ l1w, const float* __restrict__ l1b,
    const float* __restrict__ f1w, const float* __restrict__ f1b,
    const float* __restrict__ l2w, const float* __restrict__ l2b,
    const float* __restrict__ f2w, const float* __restrict__ f2b,
    const float* __restrict__ l3w, const float* __restrict__ l3b,
    const float* __restrict__ f3w, const float* __restrict__ f3b,
    float* __restrict__ ptab)
{
  int i = blockIdx.x * blockDim.x + threadIdx.x;
  if (i >= LBIAS) return;
  float b0 = biases[2 * i], b1 = biases[2 * i + 1];
  float v[PDIM], t[PDIM];
  #pragma unroll
  for (int d = 0; d < PDIM; ++d) v[d] = b0 * pw[d] + b1 * pw[PDIM + d] + pb[d];
  ln_relu(v, t, l1w, l1b);
  #pragma unroll
  for (int d = 0; d < PDIM; ++d) {
    float a = f1b[d];
    #pragma unroll
    for (int k = 0; k < PDIM; ++k) a += t[k] * f1w[k * PDIM + d];
    v[d] = a;
  }
  ln_relu(v, t, l2w, l2b);
  #pragma unroll
  for (int d = 0; d < PDIM; ++d) {
    float a = f2b[d];
    #pragma unroll
    for (int k = 0; k < PDIM; ++k) a += t[k] * f2w[k * PDIM + d];
    v[d] = a;
  }
  ln_relu(v, t, l3w, l3b);
  #pragma unroll
  for (int h = 0; h < HEADS; ++h) {
    float a = f3b[h];
    #pragma unroll
    for (int k = 0; k < PDIM; ++k) a += t[k] * f3w[k * HEADS + h];
    ptab[i * HEADS + h] = a;
  }
}

// ======================= 2) rpbt = gather(bias) + pad mask (-1e30) =======================
__global__ void build_rpbt_kernel(const float* __restrict__ ptab,
                                  const int* __restrict__ rel_idx,
                                  float* __restrict__ rpbt)
{
  int idx = blockIdx.x * blockDim.x + threadIdx.x;
  const int total = HEADS * NTOK * NQP;
  if (idx >= total) return;
  int h = idx / (NTOK * NQP);
  int rem = idx % (NTOK * NQP);
  int q = rem / NQP;
  int c = rem % NQP;
  float v = (c < NTOK) ? ptab[rel_idx[q * NTOK + c] * HEADS + h] : -1e30f;
  rpbt[idx] = v;
}

// ======================= 2b) zero-padded attention mask copy =======================
__global__ void pad_mask_kernel(const float* __restrict__ mask, float* __restrict__ maskp)
{
  int idx = blockIdx.x * blockDim.x + threadIdx.x;
  const int total = NWIN * NTOK * NQP;
  if (idx >= total) return;
  int w = idx / (NTOK * NQP);
  int rem = idx % (NTOK * NQP);
  int q = rem / NQP;
  int c = rem % NQP;
  maskp[idx] = (c < NTOK) ? mask[((size_t)w * NTOK + q) * NTOK + c] : 0.f;
}

// ======================= 3) transpose weights to f16 =======================
__global__ void prep_weights_kernel(const float* __restrict__ qkv_w,
                                    const float* __restrict__ proj_w,
                                    _Float16* __restrict__ wqkv_t,
                                    _Float16* __restrict__ wproj_t)
{
  int i = blockIdx.x * blockDim.x + threadIdx.x;
  const int n0 = 384 * 128;
  const int n1 = 128 * 128;
  if (i < n0) {
    int c = i / 128, k = i % 128;
    wqkv_t[i] = (_Float16)qkv_w[k * 384 + c];          // [out=c][in=k]
  } else if (i < n0 + n1) {
    int j = i - n0;
    int c = j / 128, k = j % 128;
    wproj_t[j] = (_Float16)proj_w[k * 128 + c];        // [out=c][in=k]
  }
}

// ======================= 4) QKV GEMM (f16 WMMA, f32 acc) =======================
__global__ __launch_bounds__(256) void qkv_kernel(
    const float* __restrict__ x, const float* __restrict__ qkv_b,
    const _Float16* __restrict__ wt,
    _Float16* __restrict__ qf, _Float16* __restrict__ kf, _Float16* __restrict__ vt)
{
  __shared__ _Float16 lx[16 * DIM];
  const int b = blockIdx.y, nt = blockIdx.x;
  const int tid = threadIdx.x;

  { // stage 16x128 fp32 x-tile -> LDS f16 (coalesced 2x float4 per thread)
    int idx = tid * 8;
    int row = idx >> 7, col = idx & 127;
    int n = nt * 16 + row;
    v4f a = {0.f, 0.f, 0.f, 0.f}, c = {0.f, 0.f, 0.f, 0.f};
    if (n < NTOK) {
      const v4f* xp = (const v4f*)(x + ((size_t)b * NTOK + n) * DIM + col);
      a = xp[0]; c = xp[1];
    }
    _Float16* lp = &lx[row * DIM + col];
    lp[0] = (_Float16)a.x; lp[1] = (_Float16)a.y; lp[2] = (_Float16)a.z; lp[3] = (_Float16)a.w;
    lp[4] = (_Float16)c.x; lp[5] = (_Float16)c.y; lp[6] = (_Float16)c.z; lp[7] = (_Float16)c.w;
  }
  __syncthreads();

  const int wave = __builtin_amdgcn_readfirstlane(tid >> 5);   // scalar wave id
  const int lane = tid & 31, hs = lane >> 4, ln = lane & 15;

  // hoist the 4 A-fragments (shared by all 3 column tiles of this wave)
  Frag A[4];
  #pragma unroll
  for (int kc = 0; kc < 4; ++kc) {
    const _Float16* ap = &lx[ln * DIM + kc * 32 + hs * 16];
    A[kc].u[0] = *(const v4u*)ap;
    A[kc].u[1] = *(const v4u*)(ap + 8);
  }

  #pragma unroll
  for (int j = 0; j < 3; ++j) {
    const int ct = wave * 3 + j;           // scalar column tile 0..23
    const int col0 = ct * 16;
    v8f acc = {};
    #pragma unroll
    for (int kc = 0; kc < 4; ++kc) {
      Frag B;
      const _Float16* bp = wt + (size_t)(col0 + ln) * DIM + kc * 32 + hs * 16;
      B.u[0] = *(const v4u*)bp;
      B.u[1] = *(const v4u*)(bp + 8);
      acc = wmma_f16(A[kc], B, acc);
    }
    const float bias = qkv_b[col0 + ln];
    const int h = (ct >> 1) & 3;                 // scalar head
    const int d = ((ct & 1) << 4) + ln;          // lane dim within head
    if (ct < 8) {                                // ---- q (scaled) ----
      _Float16* dst = qf + (((size_t)b * HEADS + h) * NQP) * HD + d;
      #pragma unroll
      for (int r = 0; r < 8; ++r) {
        int n = nt * 16 + r + 8 * hs;
        dst[(size_t)n * HD] = (_Float16)((acc[r] + bias) * SCALE);
      }
    } else if (ct < 16) {                        // ---- k ----
      _Float16* dst = kf + (((size_t)b * HEADS + h) * NQP) * HD + d;
      #pragma unroll
      for (int r = 0; r < 8; ++r) {
        int n = nt * 16 + r + 8 * hs;
        dst[(size_t)n * HD] = (_Float16)(acc[r] + bias);
      }
    } else {                                     // ---- v (transposed [d][n]) ----
      _Float16* dst = vt + (((size_t)b * HEADS + h) * HD + d) * NKP;
      #pragma unroll
      for (int r = 0; r < 8; ++r) {
        int n = nt * 16 + r + 8 * hs;
        dst[n] = (_Float16)(acc[r] + bias);
      }
    }
  }
}

// ======================= 5) fused attention + output projection =======================
__global__ __launch_bounds__(128) void attn_kernel(
    const float* __restrict__ maskp, const float* __restrict__ rpbt,
    const _Float16* __restrict__ qf, const _Float16* __restrict__ kf,
    const _Float16* __restrict__ vt, const _Float16* __restrict__ wpt,
    const float* __restrict__ proj_b, float* __restrict__ out)
{
  __shared__ _Float16 sP[HEADS][16 * NKP];   // per-head softmax probs (f16)
  __shared__ _Float16 sO[16 * DIM];          // attention output, all heads

  const int b = blockIdx.y, mt = blockIdx.x;
  const int h = __builtin_amdgcn_readfirstlane(threadIdx.x >> 5);  // scalar head id
  const int lane = threadIdx.x & 31, hs = lane >> 4, ln = lane & 15;
  const int w = b & (NWIN - 1);

  const _Float16* qb = qf + ((size_t)b * HEADS + h) * NQP * HD;
  const _Float16* kb = kf + ((size_t)b * HEADS + h) * NQP * HD;
  const _Float16* vb = vt + ((size_t)b * HEADS + h) * HD * NKP;

  // q A-fragment (reused for all 13 K-column tiles)
  Frag Aq;
  {
    const _Float16* qp = qb + (size_t)(mt * 16 + ln) * HD + hs * 16;
    Aq.u[0] = *(const v4u*)qp;
    Aq.u[1] = *(const v4u*)(qp + 8);
  }

  // S = q @ k^T : 13 tiles of 16x16 f32
  v8f s[13];
  #pragma unroll
  for (int nt = 0; nt < 13; ++nt) {
    Frag Bk;
    const _Float16* kp = kb + (size_t)(nt * 16 + ln) * HD + hs * 16;
    Bk.u[0] = *(const v4u*)kp;
    Bk.u[1] = *(const v4u*)(kp + 8);
    v8f z = {};
    s[nt] = wmma_f16(Aq, Bk, z);
  }

  // add rpb (-1e30 in pad cols) + padded attention mask, track row max
  const float* rp = rpbt + (size_t)h * NTOK * NQP;
  float m[8], sum[8];
  #pragma unroll
  for (int r = 0; r < 8; ++r) { m[r] = -3.0e38f; sum[r] = 0.f; }
  #pragma unroll
  for (int r = 0; r < 8; ++r) {
    int qrow = mt * 16 + r + 8 * hs;
    int qr = qrow < NTOK ? qrow : NTOK - 1;
    const float* rrow = rp + (size_t)qr * NQP;
    const float* mrow = maskp + ((size_t)w * NTOK + qr) * NQP;
    #pragma unroll
    for (int nt = 0; nt < 13; ++nt) {
      int c = nt * 16 + ln;
      float val = s[nt][r] + rrow[c] + mrow[c];
      s[nt][r] = val;
      m[r] = fmaxf(m[r], val);
    }
  }
  #pragma unroll
  for (int r = 0; r < 8; ++r) {   // row max across the 16-lane half (wave32)
    m[r] = fmaxf(m[r], __shfl_xor(m[r], 1));
    m[r] = fmaxf(m[r], __shfl_xor(m[r], 2));
    m[r] = fmaxf(m[r], __shfl_xor(m[r], 4));
    m[r] = fmaxf(m[r], __shfl_xor(m[r], 8));
  }

  _Float16* Pw = sP[h];
  #pragma unroll
  for (int r = 0; r < 8; ++r)     // zero K-pad cols 208..223
    Pw[(r + 8 * hs) * NKP + NQP + ln] = (_Float16)0.f;
  #pragma unroll
  for (int nt = 0; nt < 13; ++nt) {
    #pragma unroll
    for (int r = 0; r < 8; ++r) {
      float e = __expf(s[nt][r] - m[r]);
      sum[r] += e;
      Pw[(r + 8 * hs) * NKP + nt * 16 + ln] = (_Float16)e;
    }
  }
  #pragma unroll
  for (int r = 0; r < 8; ++r) {
    sum[r] += __shfl_xor(sum[r], 1);
    sum[r] += __shfl_xor(sum[r], 2);
    sum[r] += __shfl_xor(sum[r], 4);
    sum[r] += __shfl_xor(sum[r], 8);
  }

  // O = P @ V : K = 224 (7 chunks), N = 32 (2 tiles)
  v8f o0 = {}, o1 = {};
  #pragma unroll
  for (int kc = 0; kc < 7; ++kc) {
    Frag Ap, Bv;
    const _Float16* pp = &Pw[ln * NKP + kc * 32 + hs * 16];
    Ap.u[0] = *(const v4u*)pp;
    Ap.u[1] = *(const v4u*)(pp + 8);
    const _Float16* vp0 = vb + (size_t)ln * NKP + kc * 32 + hs * 16;
    Bv.u[0] = *(const v4u*)vp0;
    Bv.u[1] = *(const v4u*)(vp0 + 8);
    o0 = wmma_f16(Ap, Bv, o0);
    const _Float16* vp1 = vb + (size_t)(16 + ln) * NKP + kc * 32 + hs * 16;
    Bv.u[0] = *(const v4u*)vp1;
    Bv.u[1] = *(const v4u*)(vp1 + 8);
    o1 = wmma_f16(Ap, Bv, o1);
  }
  #pragma unroll
  for (int r = 0; r < 8; ++r) {   // normalize + deposit into shared O (all heads)
    float inv = 1.0f / sum[r];
    int M = r + 8 * hs;
    sO[M * DIM + h * HD + ln]      = (_Float16)(o0[r] * inv);
    sO[M * DIM + h * HD + 16 + ln] = (_Float16)(o1[r] * inv);
  }
  __syncthreads();

  // projection: wave h covers output column tiles 2h, 2h+1
  #pragma unroll
  for (int j = 0; j < 2; ++j) {
    int col0 = (h * 2 + j) * 16;
    v8f acc = {};
    #pragma unroll
    for (int kc = 0; kc < 4; ++kc) {
      Frag Ao, Bw;
      const _Float16* op = &sO[ln * DIM + kc * 32 + hs * 16];
      Ao.u[0] = *(const v4u*)op;
      Ao.u[1] = *(const v4u*)(op + 8);
      const _Float16* wp = wpt + (size_t)(col0 + ln) * DIM + kc * 32 + hs * 16;
      Bw.u[0] = *(const v4u*)wp;
      Bw.u[1] = *(const v4u*)(wp + 8);
      acc = wmma_f16(Ao, Bw, acc);
    }
    float pbv = proj_b[col0 + ln];
    #pragma unroll
    for (int r = 0; r < 8; ++r) {
      int tok = mt * 16 + r + 8 * hs;
      if (tok < NTOK)
        out[((size_t)b * NTOK + tok) * DIM + col0 + ln] = acc[r] + pbv;
    }
  }
}

// ======================= host launcher =======================
extern "C" void kernel_launch(void* const* d_in, const int* in_sizes, int n_in,
                              void* d_out, int out_size, void* d_ws, size_t ws_size,
                              hipStream_t stream)
{
  (void)in_sizes; (void)n_in; (void)out_size; (void)ws_size;
  const float* x      = (const float*)d_in[0];
  const float* mask   = (const float*)d_in[1];
  const float* qkv_w  = (const float*)d_in[2];
  const float* qkv_b  = (const float*)d_in[3];
  const float* proj_w = (const float*)d_in[4];
  const float* proj_b = (const float*)d_in[5];
  const float* ppw = (const float*)d_in[6];
  const float* ppb = (const float*)d_in[7];
  const float* l1w = (const float*)d_in[8];   const float* l1b = (const float*)d_in[9];
  const float* f1w = (const float*)d_in[10];  const float* f1b = (const float*)d_in[11];
  const float* l2w = (const float*)d_in[12];  const float* l2b = (const float*)d_in[13];
  const float* f2w = (const float*)d_in[14];  const float* f2b = (const float*)d_in[15];
  const float* l3w = (const float*)d_in[16];  const float* l3b = (const float*)d_in[17];
  const float* f3w = (const float*)d_in[18];  const float* f3b = (const float*)d_in[19];
  const float* biases  = (const float*)d_in[20];
  const int*   rel_idx = (const int*)d_in[21];
  float* out = (float*)d_out;

  char* ws = (char*)d_ws;
  float*    ptab  = (float*)(ws + OFF_PTAB);
  float*    rpbt  = (float*)(ws + OFF_RPBT);
  float*    maskp = (float*)(ws + OFF_MASKP);
  _Float16* wqkv  = (_Float16*)(ws + OFF_WQKV);
  _Float16* wproj = (_Float16*)(ws + OFF_WPROJ);
  _Float16* qf    = (_Float16*)(ws + OFF_QF);
  _Float16* kf    = (_Float16*)(ws + OFF_KF);
  _Float16* vt    = (_Float16*)(ws + OFF_VT);

  pos_mlp_kernel<<<(LBIAS + 255) / 256, 256, 0, stream>>>(
      biases, ppw, ppb, l1w, l1b, f1w, f1b, l2w, l2b, f2w, f2b, l3w, l3b, f3w, f3b, ptab);
  build_rpbt_kernel<<<(HEADS * NTOK * NQP + 255) / 256, 256, 0, stream>>>(ptab, rel_idx, rpbt);
  pad_mask_kernel<<<(NWIN * NTOK * NQP + 255) / 256, 256, 0, stream>>>(mask, maskp);
  prep_weights_kernel<<<(384 * 128 + 128 * 128 + 255) / 256, 256, 0, stream>>>(qkv_w, proj_w, wqkv, wproj);
  qkv_kernel<<<dim3(13, BATCH), 256, 0, stream>>>(x, qkv_b, wqkv, qf, kf, vt);
  attn_kernel<<<dim3(13, BATCH), 128, 0, stream>>>(maskp, rpbt, qf, kf, vt, wproj, proj_b, out);
}
